// _BaseConnectedComponentLoss_29257317220479
// MI455X (gfx1250) — matrix operand order repaired and checked by reading, not compiled, not run on hardware
//
#include <hip/hip_runtime.h>

// ---------------------------------------------------------------------------
// Connected-component soft-Dice loss, CDNA5 (gfx1250, wave32).
// Segmented reduction mapped onto V_WMMA_F32_16X16X32_F16:
//   D[bin][q] += sum_k onehot[bin][voxel_k] * value[q][voxel_k]
// bins  = 2 batches x 8 components (component 0 dropped)  -> 16 rows (M)
// quant = {inter = p_true, psum ~= 1.0, count = 1.0}       -> cols 0..2 (N)
//
// Voxel <-> K map: voxel = tile*64 + 32*chunk + K. Lane L owns voxels
// {2L, 2L+1} (one B64 load per array), so every aligned K pair {2m,2m+1} of
// any A/B VGPR lives in ONE source lane (lane 16*chunk + m): one ds_bpermute
// fills a whole VGPR.
//   - mask word  = oh16(v0) | oh16(v1)<<16   (bit r <=> bin r; 0 if comp==0)
//     A VGPR     = ((g >> n) & 0x00010001) * 0x3C00   (f16 1.0 one-hot pair)
//   - pt word    = f16(pt0) | f16(pt1)<<16  (already in B pair order)
//     B VGPR     = (n==0) ? g : cword       (cols 1,2 are constant 1.0h)
// comp==0 voxels have an all-zero one-hot column -> contribute nothing.
// ---------------------------------------------------------------------------

typedef __attribute__((ext_vector_type(16))) _Float16 v16h;
typedef __attribute__((ext_vector_type(8)))  float    v8f;
typedef __attribute__((ext_vector_type(2)))  float    f32x2;
typedef __attribute__((ext_vector_type(2)))  int      i32x2;

#define NCOMP   8
#define SMOOTHF 1e-5f
#define BATCH   2
#define VVOX    4096000u               // 160*160*160
#define NVOX    (BATCH * VVOX)         // 8,192,000
#define NTILE   (NVOX / 64u)           // 128,000 tiles (exact)
#define TILEB   (VVOX / 64u)           // 64,000 tiles in batch 0 (exact)
#define BLOCKS  512
#define THREADS 256
#define WPB     (THREADS / 32)
#define PARTIAL 48                     // 16 bins * 3 quantities

static __device__ __forceinline__ unsigned f16bits(float f) {
  union { _Float16 h; unsigned short s; } c;
  c.h = (_Float16)f;
  return (unsigned)c.s;
}

// p_true = softmax(y_pred)[y]  ==  sigmoid(+-(a1-a0)); f16-rounded anyway.
static __device__ __forceinline__ float ptrue(float a0, float a1, int yv) {
  const float d = a1 - a0;
  const float s = yv ? -d : d;
  return __builtin_amdgcn_rcpf(1.0f + __expf(s));
}

__global__ void __launch_bounds__(THREADS)
cc_dice_partial(const float* __restrict__ ypred,
                const int*   __restrict__ ylab,
                const int*   __restrict__ comp,
                float*       __restrict__ partial)
{
  __shared__ float red[WPB][PARTIAL];

  const unsigned tid   = threadIdx.x;
  const unsigned lane  = tid & 31u;
  const unsigned wib   = tid >> 5;
  const unsigned wgl   = (blockIdx.x * THREADS + tid) >> 5;   // uniform per wave
  const unsigned nwave = (gridDim.x * THREADS) >> 5;

  const unsigned n  = lane & 15u;   // A row / B column / D column of this lane
  const unsigned hh = lane >> 4;    // lane half (selects K sub-range)
  // B columns: n==0 -> p_true pair, n==1 (psum) / n==2 (count) -> 1.0h, else 0.
  const unsigned cword = (n == 1u || n == 2u) ? 0x3C003C00u : 0u;

  v8f acc = {};                     // D: 16x16 f32 accumulator (8 VGPRs)

  // One K=32 WMMA chunk c: gather lane 16c+m owns K pair {2m,2m+1}.
  auto mmacc = [&](unsigned c, unsigned maskw, unsigned ptw) {
    unsigned au[8], bu[8];
    // A (16x32 f16 one-hot). VGPR j: K0 = (j<4 ? 2j : 16+2(j-4)) + 8*hh.
#pragma unroll
    for (int j = 0; j < 8; ++j) {
      const unsigned m  = ((j < 4) ? (unsigned)j : (unsigned)(j + 4)) + 4u * hh;
      const unsigned ia = (16u * c + m) << 2;
      const unsigned g  = (unsigned)__builtin_amdgcn_ds_bpermute((int)ia, (int)maskw);
      au[j] = ((g >> n) & 0x00010001u) * 0x3C00u;   // f16 1.0 in hit halves
    }
    // B (32x16 f16 values). VGPR j: K0 = (j<4 ? 2j : 16+2(j-4)) + 8*hh.
#pragma unroll
    for (int j = 0; j < 8; ++j) {
      const unsigned ib = (16u * c + ((j < 4) ? (unsigned)j : (unsigned)(j + 4)) + 4u * hh) << 2;
      const unsigned g  = (unsigned)__builtin_amdgcn_ds_bpermute((int)ib, (int)ptw);
      bu[j] = (n == 0u) ? g : cword;                // pt pair already ordered
    }
    union { unsigned u[8]; v16h h; } A, B;
#pragma unroll
    for (int j = 0; j < 8; ++j) { A.u[j] = au[j]; B.u[j] = bu[j]; }
    acc = __builtin_amdgcn_wmma_f32_16x16x32_f16(
        false, A.h, false, B.h, (short)0, acc, false, false);
  };

  for (unsigned t = wgl; t < NTILE; t += nwave) {
    const unsigned b   = (t >= TILEB) ? 1u : 0u;     // uniform per tile
    const size_t   v0  = (size_t)t * 64u + 2u * lane;    // lane's 2 voxels
    const size_t   vin = v0 - (size_t)b * VVOX;          // in-batch offset

    // Read-once streaming loads, 2 voxels per lane (B64), non-temporal.
    const f32x2 c0 = __builtin_nontemporal_load(
        (const f32x2*)(ypred + (size_t)b * 2u * VVOX + vin));
    const f32x2 c1 = __builtin_nontemporal_load(
        (const f32x2*)(ypred + (size_t)b * 2u * VVOX + VVOX + vin));
    const i32x2 yy = __builtin_nontemporal_load((const i32x2*)(ylab + v0));
    const i32x2 cc = __builtin_nontemporal_load((const i32x2*)(comp + v0));

    // Per-lane payload words (2 voxels each).
    const float pt0 = ptrue(c0.x, c1.x, yy.x);
    const float pt1 = ptrue(c0.y, c1.y, yy.y);
    const unsigned ptw = f16bits(pt0) | (f16bits(pt1) << 16);

    const unsigned sh  = 8u * b;                     // bin = comp-1 + 8*b
    const unsigned oh0 = (cc.x == 0) ? 0u : (1u << ((unsigned)cc.x - 1u + sh));
    const unsigned oh1 = (cc.y == 0) ? 0u : (1u << ((unsigned)cc.y - 1u + sh));
    const unsigned maskw = oh0 | (oh1 << 16);

    mmacc(0u, maskw, ptw);   // chunk 0: voxels tile*64 + 0..31
    mmacc(1u, maskw, ptw);   // chunk 1: voxels tile*64 + 32..63
  }

  // Deterministic block reduction: lanes with N<3 hold live columns.
  // D layout: VGPR r, lanes 0-15 -> M=r ; lanes 16-31 -> M=r+8.
#pragma unroll
  for (int r = 0; r < 8; ++r) {
    if (n < 3u) {
      const unsigned m = (unsigned)r + hh * 8u;
      red[wib][m * 3u + n] = acc[r];
    }
  }
  __syncthreads();
  if (tid < PARTIAL) {
    float s = 0.0f;
#pragma unroll
    for (int w = 0; w < WPB; ++w) s += red[w][tid];
    partial[(size_t)blockIdx.x * PARTIAL + tid] = s;   // overwrite: no init needed
  }
}

__global__ void __launch_bounds__(64)
cc_dice_final(const float* __restrict__ partial, float* __restrict__ out)
{
  __shared__ float sums[PARTIAL];
  const unsigned tid = threadIdx.x;
  if (tid < PARTIAL) {
    float s = 0.0f;
    for (int blk = 0; blk < BLOCKS; ++blk)            // fixed order: deterministic
      s += partial[(size_t)blk * PARTIAL + tid];
    sums[tid] = s;
  }
  __syncthreads();
  if (tid == 0) {
    float accb = 0.0f;
    for (int b = 0; b < BATCH; ++b) {
      float sd = 0.0f, sp = 0.0f;
      for (int c = 0; c < NCOMP; ++c) {
        const int m = b * NCOMP + c;
        const float I = sums[m * 3 + 0];
        const float P = sums[m * 3 + 1];
        const float C = sums[m * 3 + 2];   // count == tsum
        if (C > 0.0f) {
          sd += 1.0f - (2.0f * I + SMOOTHF) / (P + C + SMOOTHF);
          sp += 1.0f;
        }
      }
      accb += sd / fmaxf(sp, 1.0f);
    }
    out[0] = accb * (1.0f / (float)BATCH);
  }
}

extern "C" void kernel_launch(void* const* d_in, const int* in_sizes, int n_in,
                              void* d_out, int out_size, void* d_ws, size_t ws_size,
                              hipStream_t stream) {
  (void)in_sizes; (void)n_in; (void)out_size; (void)ws_size;
  const float* ypred = (const float*)d_in[0];
  const int*   ylab  = (const int*)d_in[1];
  const int*   comp  = (const int*)d_in[2];
  float* partial = (float*)d_ws;                      // BLOCKS * 48 floats = 96 KB

  cc_dice_partial<<<BLOCKS, THREADS, 0, stream>>>(ypred, ylab, comp, partial);
  cc_dice_final<<<1, 64, 0, stream>>>(partial, (float*)d_out);
}
